// WindowAttentionV2_16398185136202
// MI455X (gfx1250) — matrix operand order repaired and checked by reading, not compiled, not run on hardware
//
#include <hip/hip_runtime.h>

typedef __bf16 bf16_t;
typedef __attribute__((ext_vector_type(16))) __bf16 v16bf;
typedef __attribute__((ext_vector_type(8)))  float  v8f;
typedef __attribute__((ext_vector_type(8)))  unsigned short v8us;
typedef __attribute__((ext_vector_type(4)))  unsigned int v4u;
typedef __attribute__((ext_vector_type(8)))  int v8i;
typedef __attribute__((ext_vector_type(4)))  int v4i;

#define WMMA_BF16(a, b, c) \
  __builtin_amdgcn_wmma_f32_16x16x32_bf16(false, (a), false, (b), (short)0, (c), false, false)

#if __has_builtin(__builtin_amdgcn_tensor_load_to_lds)
#define HAVE_TDM 1
#else
#define HAVE_TDM 0
#endif

union frag_u { v16bf v; v8us u[2]; };

// load a 16-element bf16 fragment from two 16B-aligned chunks (ds/global _b128 x2)
static __device__ __forceinline__ v16bf load_frag2(const bf16_t* p0, const bf16_t* p1) {
  frag_u t;
  t.u[0] = *(const v8us*)p0;
  t.u[1] = *(const v8us*)p1;
  return t.v;
}
// load a 16-element bf16 fragment from one contiguous 32B chunk
static __device__ __forceinline__ v16bf load_frag(const bf16_t* p) {
  return load_frag2(p, p + 8);
}

// ---------------- problem constants ----------------
// b=2048, n=64, c=512, h=16, d=32, nw=64, window 4x4x4, CPB hidden 512, table 343

// ---------------- weight conversion fp32 -> bf16 in WMMA B-fragment order ----
// fragment layout: tile = colTile*16 + kt ; within tile: lane*16 + e
//   element = W[kt*32 + (lane>=16?16:0) + e][colTile*16 + (lane&15)]
__global__ void k_cvt(const float* __restrict__ qkvw, const float* __restrict__ projw,
                      bf16_t* __restrict__ qo, bf16_t* __restrict__ po) {
  const int stride = gridDim.x * blockDim.x;
  const int i0 = blockIdx.x * blockDim.x + threadIdx.x;
  for (int f = i0; f < 512 * 1536; f += stride) {
    int tile = f >> 9, lane = (f >> 4) & 31, e = f & 15;
    int ct = tile >> 4, kt = tile & 15;
    int row = kt * 32 + ((lane >= 16) ? 16 : 0) + e;
    int col = ct * 16 + (lane & 15);
    qo[f] = (bf16_t)qkvw[row * 1536 + col];
  }
  for (int f = i0; f < 512 * 512; f += stride) {
    int tile = f >> 9, lane = (f >> 4) & 31, e = f & 15;
    int ct = tile >> 4, kt = tile & 15;
    int row = kt * 32 + ((lane >= 16) ? 16 : 0) + e;
    int col = ct * 16 + (lane & 15);
    po[f] = (bf16_t)projw[row * 512 + col];
  }
}

// ---------------- CPB MLP: bias_table[343][16] ----------------
__global__ void k_cpb(const float* __restrict__ w1, const float* __restrict__ b1,
                      const float* __restrict__ w2, float* __restrict__ table) {
  int t = blockIdx.x * blockDim.x + threadIdx.x;
  if (t >= 343 * 16) return;
  int row = t >> 4, hd = t & 15;
  int ci = row / 49, cj = (row / 7) % 7, ck = row % 7;
  float c[3] = { (float)(ci - 3), (float)(cj - 3), (float)(ck - 3) };
  const float inv_log2_8 = 1.0f / log2f(8.0f);
#pragma unroll
  for (int a = 0; a < 3; ++a) {
    float v = c[a] * (8.0f / 3.0f);
    float s = (v > 0.f) ? 1.f : ((v < 0.f) ? -1.f : 0.f);
    c[a] = s * log2f(fabsf(v) + 1.0f) * inv_log2_8;
  }
  float acc = 0.f;
  for (int j = 0; j < 512; ++j) {
    float hsum = c[0] * w1[j] + c[1] * w1[512 + j] + c[2] * w1[1024 + j] + b1[j];
    hsum = fmaxf(hsum, 0.f);
    acc += hsum * w2[j * 16 + hd];
  }
  table[row * 16 + hd] = acc;
}

// ---------------- bias expansion: 16*sigmoid(table[rpi]) -> [16][64][64] ------
__global__ void k_bias(const float* __restrict__ table, float* __restrict__ bias) {
  int t = blockIdx.x * blockDim.x + threadIdx.x;
  if (t >= 16 * 64 * 64) return;
  int hd = t >> 12, i = (t >> 6) & 63, j = t & 63;
  int di = i >> 4, hi = (i >> 2) & 3, wi = i & 3;
  int dj = j >> 4, hj = (j >> 2) & 3, wj = j & 3;
  int idx = (di - dj + 3) * 49 + (hi - hj + 3) * 7 + (wi - wj + 3);
  float v = table[idx * 16 + hd];
  bias[t] = 16.0f / (1.0f + expf(-v));
}

// ---------------- fully fused: QKV + cosine attention + projection ----------
__global__ __launch_bounds__(256, 1)
void k_attn(const float* __restrict__ x, const float* __restrict__ mask,
            const bf16_t* __restrict__ qkvw, const float* __restrict__ qkvb,
            const float* __restrict__ lscale, const float* __restrict__ bias16,
            const bf16_t* __restrict__ projw, const float* __restrict__ projb,
            float* __restrict__ out) {
  __shared__ __align__(16) bf16_t xs[64 * 512];       // 64KB x tile (row major)
  __shared__ __align__(16) bf16_t ys[64 * 512];       // 64KB attn output (also TDM bounce buf)
  __shared__ __align__(16) bf16_t qb[64 * 32];        // normalized q (A layout)
  __shared__ __align__(16) bf16_t kfrag[4 * 32 * 16]; // kn^T in B-fragment order
  __shared__ __align__(16) bf16_t vfrag[4 * 32 * 16]; // V in B-fragment order
  // lifetime-overlaid region: {qf,kf} during QKV/normalize, then sS (+ in-place pbuf)
  union RegA {
    struct { float qf[64 * 32]; float kf[64 * 32]; } s;
    float sS[64 * 64];
  };
  __shared__ __align__(16) RegA regA;                 // 16KB
  bf16_t* const pbuf = (bf16_t*)regA.sS;              // probs in place, row stride 128

  const int tid = threadIdx.x;
  const int wave = tid >> 5, lane = tid & 31;
  const int ln15 = lane & 15;
  const int hiHalf = (lane >= 16) ? 1 : 0;
  const int koffA = hiHalf ? 16 : 0;                  // 16-bit A-matrix lane K offset
  const int bwin = blockIdx.x;
  const int wm = bwin & 63;

#if HAVE_TDM
  // ---- stage x via Tensor Data Mover, 2 halves of 32 rows (64KB fp32 each) ----
  {
    float* const xstage = (float*)ys;                 // ys is dead until first P@V
    for (int half = 0; half < 2; ++half) {
      if (wave == 0) {
        unsigned long long ga =
            (unsigned long long)(size_t)(x + (size_t)bwin * 64 * 512 + (size_t)half * 32 * 512);
        v4u g0;
        g0[0] = 1u;                                   // count=1, user descriptor
        g0[1] = (unsigned)(size_t)xstage;             // lds_addr
        g0[2] = (unsigned)(ga & 0xFFFFFFFFu);         // global_addr[31:0]
        g0[3] = (unsigned)((ga >> 32) & 0x01FFFFFFu)  // global_addr[56:32]
              | (2u << 30);                           // type=2 ("image")
        v8i g1 = 0;
        g1[0] = 2 << 16;                              // data_size=2 (4B), wg_mask=0
        g1[1] = 512 << 16;                            // tensor_dim0 = 512 (lo16 @ bit48)
        g1[2] = 32 << 16;                             // dim0 hi16=0 | tensor_dim1 = 32
        g1[3] = 512 << 16;                            // dim1 hi16=0 | tile_dim0 = 512
        g1[4] = 32;                                   // tile_dim1 = 32, tile_dim2 = 0
        g1[5] = 512;                                  // tensor_dim0_stride = 512 (48b)
        v4i z4 = 0;
        v8i z8 = 0;
        __builtin_amdgcn_tensor_load_to_lds(g0, g1, z4, z4, z8, 0);
#if __has_builtin(__builtin_amdgcn_s_wait_tensorcnt)
        __builtin_amdgcn_s_wait_tensorcnt(0);
#else
        asm volatile("s_wait_tensorcnt 0x0" ::: "memory");
#endif
      }
      __syncthreads();
      // convert the staged 32 rows fp32 -> bf16 into xs
      for (int idx = tid; idx < 32 * 512 / 4; idx += 256) {
        float4 v = ((const float4*)xstage)[idx];
        int o = half * 32 * 512 + idx * 4;
        xs[o + 0] = (bf16_t)v.x; xs[o + 1] = (bf16_t)v.y;
        xs[o + 2] = (bf16_t)v.z; xs[o + 3] = (bf16_t)v.w;
      }
      __syncthreads();
    }
  }
#else
  // ---- fallback: stage x tile through VGPRs (fp32 -> bf16 LDS) ----
  const float4* xg = (const float4*)(x + (size_t)bwin * 64 * 512);
  for (int idx = tid; idx < 64 * 512 / 4; idx += 256) {
    float4 v = xg[idx];
    int o = idx * 4;
    xs[o + 0] = (bf16_t)v.x; xs[o + 1] = (bf16_t)v.y;
    xs[o + 2] = (bf16_t)v.z; xs[o + 3] = (bf16_t)v.w;
  }
  __syncthreads();
#endif

  const int mt = wave >> 1, nt = wave & 1;            // 4x2 tiling of 64x32 per-head outputs
  const int Arow = mt * 16 + ln15;

  for (int h = 0; h < 16; ++h) {
    float scale_h = expf(fminf(lscale[h], 4.60517018599f));  // min(ls, ln(100))

    // ---- Q/K/V = x @ W (K=512, 16 WMMA steps), one 16x16 tile per wave ----
    for (int r = 0; r < 3; ++r) {
      const int colTile = r * 32 + h * 2 + nt;        // 16-column tile index in qkv_w
      const bf16_t* wbase = qkvw + (size_t)colTile * 16 * 512 + lane * 16;
      v8f acc = {};
      for (int kk = 0; kk < 16; ++kk) {
        __builtin_prefetch(wbase + (kk + 1) * 512, 0, 1);
        v16bf a = load_frag2(&xs[Arow * 512 + kk * 32 + koffA],
                             &xs[Arow * 512 + kk * 32 + 8 + koffA]);
        v16bf b = load_frag(wbase + kk * 512);        // 32B contiguous per lane
        acc = WMMA_BF16(a, b, acc);
      }
      float badd = qkvb[r * 512 + h * 32 + nt * 16 + ln15];
#pragma unroll
      for (int i = 0; i < 8; ++i) {
        int Mg = mt * 16 + i + (hiHalf ? 8 : 0);
        int col = nt * 16 + ln15;
        float vv = acc[i] + badd;
        if (r == 0)      regA.s.qf[Mg * 32 + col] = vv;
        else if (r == 1) regA.s.kf[Mg * 32 + col] = vv;
        else {
          // scatter V directly into B-fragment order for the P@V GEMM
          int kchunk = Mg >> 5, r5 = Mg & 31;
          int lslot = ln15 + ((r5 >= 16) ? 16 : 0);
          int e = r5 & 15;
          vfrag[((nt * 2 + kchunk) * 32 + lslot) * 16 + e] = (bf16_t)vv;
        }
      }
    }
    __syncthreads();

    // ---- cosine normalization of q, k rows (d = 32) ----
    if (tid < 128) {
      int row = tid & 63;
      const float* src = (tid < 64) ? regA.s.qf : regA.s.kf;
      float vals[32];
      float ss = 0.f;
#pragma unroll
      for (int e = 0; e < 32; ++e) { vals[e] = src[row * 32 + e]; ss += vals[e] * vals[e]; }
      float rn = 1.0f / fmaxf(sqrtf(ss), 1e-12f);
      if (tid < 64) {
#pragma unroll
        for (int e = 0; e < 32; ++e) qb[row * 32 + e] = (bf16_t)(vals[e] * rn);
      } else {
        // write kn^T straight into B-fragment order (two contiguous 32B chunks)
        int snt = row >> 4, rl = row & 15;
        bf16_t* c0 = &kfrag[(snt * 32 + rl) * 16];        // K = 0..15
        bf16_t* c1 = &kfrag[(snt * 32 + 16 + rl) * 16];   // K = 16..31
#pragma unroll
        for (int e = 0; e < 16; ++e) c0[e] = (bf16_t)(vals[e] * rn);
#pragma unroll
        for (int e = 0; e < 16; ++e) c1[e] = (bf16_t)(vals[16 + e] * rn);
      }
    }
    __syncthreads();

    // ---- S = qn @ kn^T (64x64, K=32): 16 tiles, 2 per wave ----
    for (int tt = wave; tt < 16; tt += 8) {
      int smt = tt >> 2, snt = tt & 3;
      v16bf a = load_frag2(&qb[(smt * 16 + ln15) * 32 + koffA],
                           &qb[(smt * 16 + ln15) * 32 + 8 + koffA]);
      v16bf b = load_frag(&kfrag[(snt * 32 + lane) * 16]);
      v8f acc = {};
      acc = WMMA_BF16(a, b, acc);
#pragma unroll
      for (int i = 0; i < 8; ++i) {
        int Mg = smt * 16 + i + (hiHalf ? 8 : 0);
        int Ng = snt * 16 + ln15;
        float s = acc[i] * scale_h
                + bias16[h * 4096 + Mg * 64 + Ng]
                + mask[(size_t)wm * 4096 + Mg * 64 + Ng];
        regA.sS[Mg * 64 + Ng] = s;
      }
    }
    __syncthreads();

    // ---- row softmax (one row per thread); probs written in place as bf16 ----
    if (tid < 64) {
      float m = -3.4e38f;
      for (int j = 0; j < 64; ++j) m = fmaxf(m, regA.sS[tid * 64 + j]);
      float sum = 0.f;
      for (int j = 0; j < 64; ++j) {
        float e = expf(regA.sS[tid * 64 + j] - m);
        regA.sS[tid * 64 + j] = e;
        sum += e;
      }
      float rs = 1.0f / sum;
      // in-place: bf16 write at byte 2j trails fp32 read at byte 4j (same row owner)
      for (int j = 0; j < 64; ++j) {
        float e = regA.sS[tid * 64 + j];
        pbuf[tid * 128 + j] = (bf16_t)(e * rs);
      }
    }
    __syncthreads();

    // ---- O = P @ V (64x32, K=64): one 16x16 tile per wave -> ys accumulator ----
    {
      v8f acc = {};
#pragma unroll
      for (int kc = 0; kc < 2; ++kc) {
        v16bf a = load_frag2(&pbuf[Arow * 128 + kc * 32 + koffA],
                             &pbuf[Arow * 128 + kc * 32 + 8 + koffA]);
        v16bf b = load_frag(&vfrag[((nt * 2 + kc) * 32 + lane) * 16]);
        acc = WMMA_BF16(a, b, acc);
      }
#pragma unroll
      for (int i = 0; i < 8; ++i) {
        int Mg = mt * 16 + i + (hiHalf ? 8 : 0);
        ys[Mg * 512 + h * 32 + nt * 16 + ln15] = (bf16_t)acc[i];
      }
    }
    __syncthreads();
  }

  // ---- fused projection: out = ys @ proj_w + proj_b ----
  for (int t = wave; t < 128; t += 8) {               // 4 Mtiles x 32 Ntiles
    int pmt = t >> 5, pnt = t & 31;
    const bf16_t* wbase = projw + (size_t)pnt * 16 * 512 + lane * 16;
    v8f acc = {};
    for (int kk = 0; kk < 16; ++kk) {
      __builtin_prefetch(wbase + (kk + 1) * 512, 0, 1);
      v16bf a = load_frag2(&ys[(pmt * 16 + ln15) * 512 + kk * 32 + koffA],
                           &ys[(pmt * 16 + ln15) * 512 + kk * 32 + 8 + koffA]);
      v16bf b = load_frag(wbase + kk * 512);
      acc = WMMA_BF16(a, b, acc);
    }
    float badd = projb[pnt * 16 + ln15];
#pragma unroll
    for (int i = 0; i < 8; ++i) {
      int Mg = pmt * 16 + i + (hiHalf ? 8 : 0);
      out[((size_t)bwin * 64 + Mg) * 512 + pnt * 16 + ln15] = acc[i] + badd;
    }
  }
}

// ---------------- launcher ----------------
extern "C" void kernel_launch(void* const* d_in, const int* in_sizes, int n_in,
                              void* d_out, int out_size, void* d_ws, size_t ws_size,
                              hipStream_t stream) {
  (void)in_sizes; (void)n_in; (void)out_size; (void)ws_size;
  const float* x     = (const float*)d_in[0];
  const float* mask  = (const float*)d_in[1];
  const float* qkvw  = (const float*)d_in[2];
  const float* qkvb  = (const float*)d_in[3];
  const float* projw = (const float*)d_in[4];
  const float* projb = (const float*)d_in[5];
  const float* lsc   = (const float*)d_in[6];
  const float* w1    = (const float*)d_in[7];
  const float* b1    = (const float*)d_in[8];
  const float* w2    = (const float*)d_in[9];

  char* ws = (char*)d_ws;
  bf16_t* qkvw_bf = (bf16_t*)(ws);                    // 512*1536*2 = 1,572,864 B (frag order)
  bf16_t* projw_bf = (bf16_t*)(ws + 1572864);         //   512*512*2 =   524,288 B (frag order)
  float*  cpb_tab  = (float*)(ws + 2097152);          //   343*16*4  =    21,952 B
  float*  bias16   = (float*)(ws + 2129920);          // 16*64*64*4  =   262,144 B
  float* out = (float*)d_out;

  hipLaunchKernelGGL(k_cvt,  dim3(512), dim3(256), 0, stream, qkvw, projw, qkvw_bf, projw_bf);
  hipLaunchKernelGGL(k_cpb,  dim3((343 * 16 + 255) / 256), dim3(256), 0, stream, w1, b1, w2, cpb_tab);
  hipLaunchKernelGGL(k_bias, dim3((16 * 64 * 64) / 256), dim3(256), 0, stream, cpb_tab, bias16);
  hipLaunchKernelGGL(k_attn, dim3(2048), dim3(256), 0, stream,
                     x, mask, qkvw_bf, qkvb, lsc, bias16, projw_bf, projb, out);
}
